// acLSTM_70411693850648
// MI455X (gfx1250) — matrix operand order repaired and checked by reading, not compiled, not run on hardware
//
#include <hip/hip_runtime.h>

// ---------------------------------------------------------------------------
// acLSTM for MI455X (gfx1250): bf16 WMMA gate GEMMs fused with the LSTM cell
// update (gate-interleaved weights, LDS-staged accumulators, bias baked into
// an extra GEMM column, double-buffered activations). 2 launches per step.
// ---------------------------------------------------------------------------

typedef __attribute__((ext_vector_type(16))) __bf16 v16bf;
typedef __attribute__((ext_vector_type(8)))  __bf16 v8bf;
typedef __attribute__((ext_vector_type(8)))  float  v8f;

#define HID   1024
#define INF   175
#define BATCH 128
#define TLEN  200
#define K1    1216   // layer1 act row: [h0 1024 | in_frame 175 | 1.0 @1199 | 0 pad]
#define KB1   1199
#define K23   2080   // layers 2/3 act row: [x 1024 | h 1024 | 1.0 @2048 | 0 pad]
#define KB23  2048
#define KD    1056   // decoder act row: [h2 1024 | 1.0 @1024 | 0 pad]
#define NG    4096   // gate columns per layer; column n = 4*j + gate

// 64KB LDS tile, XOR-swizzled: bank = (row ^ col) & 63 -> conflict-free writes.
#define SWZ(col, row) (((col) << 7) | (((row) ^ ((col) & 63)) & 127))

__device__ __forceinline__ __bf16 f2bf(float f) {
  unsigned u = __builtin_bit_cast(unsigned, f);
  unsigned r = u + 0x7fffu + ((u >> 16) & 1u);   // round-to-nearest-even
  unsigned short h = (unsigned short)(r >> 16);
  return __builtin_bit_cast(__bf16, h);
}
__device__ __forceinline__ float sigmoidf_(float x) {
  return 1.0f / (1.0f + __expf(-x));
}

// ---------------------------------------------------------------------------
// Fused gates GEMM + cell update.
// grid.x = 3 layers * 32 N-blocks (128 gate-columns = 32 hidden units each).
// Phase 1: each of 8 waves computes a 128(M) x 16(N) tile via WMMA.
// Phase 2: accumulators -> LDS (swizzled), barrier.
// Phase 3: 4096 cell updates (b, j): read i,f,g,o from LDS, update c (f32,
//          in place), write new h (bf16) into next-step activation buffers.
// ---------------------------------------------------------------------------
__global__ void __launch_bounds__(256) fused_step_kernel(
    const __bf16* __restrict__ xh1r, const __bf16* __restrict__ xh2r,
    const __bf16* __restrict__ xh3r,
    const __bf16* __restrict__ W1,  const __bf16* __restrict__ W2,
    const __bf16* __restrict__ W3,
    float* __restrict__ cbuf,
    __bf16* __restrict__ xh1w, __bf16* __restrict__ xh2w,
    __bf16* __restrict__ xh3w, __bf16* __restrict__ dec_in)
{
  __shared__ float smem[128 * 128];   // 64 KB

  int layer = blockIdx.x >> 5;
  int nb    = blockIdx.x & 31;
  const __bf16* act; const __bf16* wgt; int K;
  if (layer == 0)      { act = xh1r; wgt = W1; K = K1;  }
  else if (layer == 1) { act = xh2r; wgt = W2; K = K23; }
  else                 { act = xh3r; wgt = W3; K = K23; }

  int tid   = threadIdx.x;
  int lane  = tid & 31;
  int wave  = tid >> 5;
  int row0  = lane & 15;
  int nloc  = wave * 16 + row0;       // local column 0..127
  int n     = nb * 128 + nloc;        // global gate column
  int kofsA = (lane >> 4) << 4;       // 0 or 16
  int kofsB = (lane >> 4) << 4;       // 0 or 16

  // ---- Phase 1: GEMM ----
  v8f acc[8];
  #pragma unroll
  for (int m = 0; m < 8; ++m) acc[m] = (v8f){0,0,0,0,0,0,0,0};

  const __bf16* wrow = wgt + (size_t)n * K;
  int nkt = K >> 5;
  for (int kt = 0; kt < nkt; ++kt) {
    int kb = kt << 5;
    v16bf bfrag = *(const v16bf*)(wrow + kb + kofsB);
    #pragma unroll
    for (int m = 0; m < 8; ++m) {
      const __bf16* ap = act + (size_t)(m * 16 + row0) * K + kb + kofsA;
      v8bf lo = *(const v8bf*)(ap);
      v8bf hi = *(const v8bf*)(ap + 8);
      v16bf afrag;
      #pragma unroll
      for (int i = 0; i < 8; ++i) { afrag[i] = lo[i]; afrag[i + 8] = hi[i]; }
      acc[m] = __builtin_amdgcn_wmma_f32_16x16x32_bf16(
          false, afrag, false, bfrag, (short)0, acc[m], false, false);
    }
  }

  // ---- Phase 2: accumulators -> LDS ----
  int hi8 = (lane >> 4) << 3;
  #pragma unroll
  for (int m = 0; m < 8; ++m) {
    #pragma unroll
    for (int r = 0; r < 8; ++r) {
      smem[SWZ(nloc, m * 16 + hi8 + r)] = acc[m][r];
    }
  }
  __syncthreads();

  // ---- Phase 3: cell update for 32 hidden units x 128 batch ----
  for (int it = 0; it < 16; ++it) {
    int idx    = it * 256 + tid;      // 0..4095
    int jl     = idx & 31;            // hidden unit within block
    int b      = idx >> 5;            // batch row
    int j      = nb * 32 + jl;        // global hidden unit
    int c0     = 4 * jl;              // local column of gate i

    float gi = smem[SWZ(c0 + 0, b)];  // bias already baked in via GEMM
    float gf = smem[SWZ(c0 + 1, b)];
    float gg = smem[SWZ(c0 + 2, b)];
    float go = smem[SWZ(c0 + 3, b)];

    float* cp = cbuf + (size_t)layer * BATCH * HID + (size_t)b * HID + j;
    float c  = *cp;
    float cn = sigmoidf_(gf) * c + sigmoidf_(gi) * tanhf(gg);
    float h  = sigmoidf_(go) * tanhf(cn);
    *cp = cn;

    __bf16 hb = f2bf(h);
    if (layer == 0) {          // nh0 -> layer1 h-slot, layer2 x-slot
      xh1w[(size_t)b * K1 + j]         = hb;
      xh2w[(size_t)b * K23 + j]        = hb;
    } else if (layer == 1) {   // nh1 -> layer2 h-slot, layer3 x-slot
      xh2w[(size_t)b * K23 + 1024 + j] = hb;
      xh3w[(size_t)b * K23 + j]        = hb;
    } else {                   // nh2 -> layer3 h-slot, decoder input
      xh3w[(size_t)b * K23 + 1024 + j] = hb;
      dec_in[(size_t)b * KD + j]       = hb;
    }
  }
}

// ---------------------------------------------------------------------------
// Decoder: out[128][175] = dec_in[128][1056] * Wdec[176][1056]^T (bias baked
// into column 1024). grid = 11 N-tiles; wave w = M-tile w. Also writes the
// next step's in_frame (cond ? x[t+1] : out) into xh1 of the NEXT buffer.
// ---------------------------------------------------------------------------
__global__ void __launch_bounds__(256) decoder_kernel(
    const __bf16* __restrict__ dec_in, const __bf16* __restrict__ Wd,
    float* __restrict__ out, const float* __restrict__ rs,
    __bf16* __restrict__ xh1next,
    int t, int cond_next, int do_feedback)
{
  int lane  = threadIdx.x & 31;
  int wave  = threadIdx.x >> 5;
  int row0  = lane & 15;
  int n     = blockIdx.x * 16 + row0;
  int kofsA = (lane >> 4) << 4;
  int kofsB = (lane >> 4) << 4;

  v8f acc = (v8f){0,0,0,0,0,0,0,0};
  const __bf16* wrow = Wd + (size_t)n * KD;
  const __bf16* arow = dec_in + (size_t)(wave * 16 + row0) * KD;

  for (int kt = 0; kt < (KD >> 5); ++kt) {
    int kb = kt << 5;
    v16bf bfrag = *(const v16bf*)(wrow + kb + kofsB);
    v8bf lo = *(const v8bf*)(arow + kb + kofsA);
    v8bf hi = *(const v8bf*)(arow + kb + kofsA + 8);
    v16bf afrag;
    #pragma unroll
    for (int i = 0; i < 8; ++i) { afrag[i] = lo[i]; afrag[i + 8] = hi[i]; }
    acc = __builtin_amdgcn_wmma_f32_16x16x32_bf16(
        false, afrag, false, bfrag, (short)0, acc, false, false);
  }

  if (n < INF) {
    int hi8 = (lane >> 4) << 3;
    #pragma unroll
    for (int r = 0; r < 8; ++r) {
      int   b = wave * 16 + hi8 + r;
      float v = acc[r];
      out[(size_t)b * (TLEN * INF) + (size_t)t * INF + n] = v;
      if (do_feedback) {
        float nf = cond_next
                 ? rs[(size_t)b * (TLEN * INF) + (size_t)(t + 1) * INF + n]
                 : v;
        xh1next[(size_t)b * K1 + 1024 + n] = f2bf(nf);
      }
    }
  }
}

// ---------------------------------------------------------------------------
// Weight prepack with gate interleaving: dst column n <- source row
// (n&3)*1024 + (n>>2).  Row layout: [A aC | B bC | bias @kbias | 0 pad].
// ---------------------------------------------------------------------------
__global__ void pack_cat_kernel(__bf16* __restrict__ dst, int K,
                                const float* __restrict__ A, int aC,
                                const float* __restrict__ Bm, int bC,
                                const float* __restrict__ bih,
                                const float* __restrict__ bhh,
                                int kbias, long total)
{
  long idx = (long)blockIdx.x * blockDim.x + threadIdx.x;
  if (idx >= total) return;
  int n  = (int)(idx / K);
  int k  = (int)(idx % K);
  int sn = (n & 3) * HID + (n >> 2);   // gate-interleave source row
  float v;
  if (k < aC)            v = A[(size_t)sn * aC + k];
  else if (k < aC + bC)  v = Bm[(size_t)sn * bC + (k - aC)];
  else if (k == kbias)   v = bih[sn] + bhh[sn];
  else                   v = 0.0f;
  dst[idx] = f2bf(v);
}

__global__ void pack_dec_kernel(__bf16* __restrict__ dst,
                                const float* __restrict__ W,
                                const float* __restrict__ bdec)
{
  int idx = blockIdx.x * blockDim.x + threadIdx.x;   // < 176*1056
  int n = idx / KD;
  int k = idx % KD;
  float v = 0.0f;
  if (n < INF) {
    if (k < HID)        v = W[(size_t)n * HID + k];
    else if (k == HID)  v = bdec[n];
  }
  dst[idx] = f2bf(v);
}

// ---------------------------------------------------------------------------
// State init: zero both activation buffers, c, dec_in.
// ---------------------------------------------------------------------------
__global__ void init_kernel(__bf16* x1a, __bf16* x1b, __bf16* x2a, __bf16* x2b,
                            __bf16* x3a, __bf16* x3b, float* cbuf,
                            __bf16* dec_in)
{
  int i = blockIdx.x * blockDim.x + threadIdx.x;   // < 3*128*1024
  __bf16 z = f2bf(0.0f);
  if (i < BATCH * K1)    { x1a[i] = z; x1b[i] = z; }
  if (i < BATCH * K23)   { x2a[i] = z; x2b[i] = z; x3a[i] = z; x3b[i] = z; }
  if (i < 3 * BATCH * HID) cbuf[i] = 0.0f;
  if (i < BATCH * KD)      dec_in[i] = z;
}

// Set constant-1.0 bias columns (both buffers) and in_frame(0) = x_0.
__global__ void seed_kernel(__bf16* x1a, __bf16* x1b, __bf16* x2a, __bf16* x2b,
                            __bf16* x3a, __bf16* x3b, __bf16* dec_in,
                            const float* rs)
{
  int i = blockIdx.x * blockDim.x + threadIdx.x;
  __bf16 one = f2bf(1.0f);
  if (i < BATCH) {
    x1a[(size_t)i * K1  + KB1]  = one;  x1b[(size_t)i * K1  + KB1]  = one;
    x2a[(size_t)i * K23 + KB23] = one;  x2b[(size_t)i * K23 + KB23] = one;
    x3a[(size_t)i * K23 + KB23] = one;  x3b[(size_t)i * K23 + KB23] = one;
    dec_in[(size_t)i * KD + HID] = one;
  }
  if (i < BATCH * INF) {
    int b = i / INF, f = i % INF;     // cond[0] is true -> in_frame(0) = x_0
    x1a[(size_t)b * K1 + 1024 + f] = f2bf(rs[(size_t)b * (TLEN * INF) + f]);
  }
}

// ---------------------------------------------------------------------------
extern "C" void kernel_launch(void* const* d_in, const int* in_sizes, int n_in,
                              void* d_out, int out_size, void* d_ws,
                              size_t ws_size, hipStream_t stream)
{
  (void)in_sizes; (void)n_in; (void)out_size; (void)ws_size;

  const float* rs   = (const float*)d_in[0];
  const float* Wih1 = (const float*)d_in[1];
  const float* Whh1 = (const float*)d_in[2];
  const float* bih1 = (const float*)d_in[3];
  const float* bhh1 = (const float*)d_in[4];
  const float* Wih2 = (const float*)d_in[5];
  const float* Whh2 = (const float*)d_in[6];
  const float* bih2 = (const float*)d_in[7];
  const float* bhh2 = (const float*)d_in[8];
  const float* Wih3 = (const float*)d_in[9];
  const float* Whh3 = (const float*)d_in[10];
  const float* bih3 = (const float*)d_in[11];
  const float* bhh3 = (const float*)d_in[12];
  const float* Wdec = (const float*)d_in[13];
  const float* bdec = (const float*)d_in[14];
  float* out = (float*)d_out;

  char* ws = (char*)d_ws;
  size_t off = 0;
  auto alloc = [&](size_t bytes) -> void* {
    off = (off + 255) & ~(size_t)255;
    void* p = ws + off;
    off += bytes;
    return p;
  };

  __bf16* W1p = (__bf16*)alloc((size_t)NG * K1  * 2);   // ~10.0 MB
  __bf16* W2p = (__bf16*)alloc((size_t)NG * K23 * 2);   // ~17.0 MB
  __bf16* W3p = (__bf16*)alloc((size_t)NG * K23 * 2);   // ~17.0 MB
  __bf16* Wdp = (__bf16*)alloc((size_t)176 * KD * 2);   // ~0.37 MB
  __bf16* xh1[2], *xh2[2], *xh3[2];
  for (int s = 0; s < 2; ++s) {
    xh1[s] = (__bf16*)alloc((size_t)BATCH * K1  * 2);
    xh2[s] = (__bf16*)alloc((size_t)BATCH * K23 * 2);
    xh3[s] = (__bf16*)alloc((size_t)BATCH * K23 * 2);
  }
  float*  cbuf   = (float*) alloc((size_t)3 * BATCH * HID * 4);
  __bf16* dec_in = (__bf16*)alloc((size_t)BATCH * KD * 2);

  // --- one-time per launch: pack weights (gate-interleaved, bias column) ---
  {
    long t1 = (long)NG * K1;
    pack_cat_kernel<<<(unsigned)((t1 + 255) / 256), 256, 0, stream>>>(
        W1p, K1, Whh1, HID, Wih1, INF, bih1, bhh1, KB1, t1);
    long t2 = (long)NG * K23;
    pack_cat_kernel<<<(unsigned)((t2 + 255) / 256), 256, 0, stream>>>(
        W2p, K23, Wih2, HID, Whh2, HID, bih2, bhh2, KB23, t2);
    pack_cat_kernel<<<(unsigned)((t2 + 255) / 256), 256, 0, stream>>>(
        W3p, K23, Wih3, HID, Whh3, HID, bih3, bhh3, KB23, t2);
    pack_dec_kernel<<<(176 * KD + 255) / 256, 256, 0, stream>>>(Wdp, Wdec, bdec);
  }

  // --- zero state, seed bias columns + in_frame(0) ---
  init_kernel<<<(3 * BATCH * HID) / 256, 256, 0, stream>>>(
      xh1[0], xh1[1], xh2[0], xh2[1], xh3[0], xh3[1], cbuf, dec_in);
  seed_kernel<<<(BATCH * INF + 255) / 256, 256, 0, stream>>>(
      xh1[0], xh1[1], xh2[0], xh2[1], xh3[0], xh3[1], dec_in, rs);

  // --- sequential scan over T=200 steps, 2 launches each ---
  for (int t = 0; t < TLEN; ++t) {
    int r = t & 1, w = r ^ 1;
    fused_step_kernel<<<96, 256, 0, stream>>>(
        xh1[r], xh2[r], xh3[r], W1p, W2p, W3p,
        cbuf, xh1[w], xh2[w], xh3[w], dec_in);
    int cond_next   = (((t + 1) % 10) < 5) ? 1 : 0;   // _condition_mask pattern
    int do_feedback = (t < TLEN - 1) ? 1 : 0;
    decoder_kernel<<<11, 256, 0, stream>>>(
        dec_in, Wdp, out, rs, xh1[w], t, cond_next, do_feedback);
  }
}
